// Net_50861002719975
// MI455X (gfx1250) — matrix operand order repaired and checked by reading, not compiled, not run on hardware
//
#include <hip/hip_runtime.h>
#include <hip/hip_bf16.h>

// ---------------------------------------------------------------------------
// Mamba-classifier forward for MI455X (gfx1250, wave32, WMMA).
// All three per-layer GEMMs run on v_wmma_f32_16x16x32_f16, with each wave
// computing a 16 x (16*NT) strip so the streamed A fragment is reused across
// NT B fragments (activation traffic is the bottleneck at 23.3 TB/s).
// ---------------------------------------------------------------------------

typedef __attribute__((ext_vector_type(16))) _Float16 v16h;
typedef __attribute__((ext_vector_type(8)))  _Float16 v8h;
typedef __attribute__((ext_vector_type(8)))  float    v8f;

#define BATCH   128
#define SEQ     512
#define HID     64
#define DI      128
#define ST      16
#define RK      4
#define KC      4
#define NLAYERS 2
#define NCLS    3
#define MROWS   (BATCH * SEQ)          // 65536

union Frag16 { v16h v; v8h h[2]; };

// ---------------------------------------------------------------------------
// f16 WMMA GEMM: C[M,N] = A[M,K] * B[N,K]^T.
// K multiple of 32, M multiple of 16, N = nGroups * NT * 16.
// One 16x(16*NT) strip per wave (A fragment reused NT times), 8 waves/block.
// mode 0: store f16 to Cf16 (ldc)
// mode 1: residual: acc initialized from Cf32, result stored back f32 (ldc)
// ---------------------------------------------------------------------------
template <int NT>
__global__ __launch_bounds__(256)
void wmma_gemm_f16(const _Float16* __restrict__ A,
                   const _Float16* __restrict__ Bw,
                   _Float16* __restrict__ Cf16,
                   float* __restrict__ Cf32,
                   int M, int K, int ldc, int nGroups, int mode)
{
    const int lane = threadIdx.x & 31;
    const int wave = threadIdx.x >> 5;
    const int t    = blockIdx.x * 8 + wave;
    const int mt   = t / nGroups;
    const int ng   = t - mt * nGroups;
    if (mt * 16 >= M) return;                 // wave-uniform exit

    const int m0  = mt << 4;
    const int n0  = ng * (16 * NT);
    const int row = lane & 15;
    const int hi  = lane >> 4;                // 0 for lanes 0-15, 1 for 16-31

    v8f acc[NT];
    #pragma unroll
    for (int j = 0; j < NT; ++j) acc[j] = (v8f){};
    if (mode == 1) {
        #pragma unroll
        for (int j = 0; j < NT; ++j)
            #pragma unroll
            for (int v = 0; v < 8; ++v)
                acc[j][v] = Cf32[(size_t)(m0 + hi * 8 + v) * ldc + (n0 + j * 16 + row)];
    }

    for (int k0 = 0; k0 < K; k0 += 32) {
        // A-matrix 16x32 f16 fragment (ISA 7.12.2): lane row = m0+row,
        // K chunks [k0 + hi*8 .. +7] and [k0 + hi*8 + 16 .. +7]
        Frag16 a;
        const _Float16* ap = A + (size_t)(m0 + row) * K + k0 + hi * 8;
        a.h[0] = *(const v8h*)(ap);
        a.h[1] = *(const v8h*)(ap + 16);

        // NT B-matrix 32x16 fragments: lane col = n0+j*16+row,
        // K chunks [k0 + hi*8 .. +7] and [k0 + hi*8 + 16 .. +7]
        Frag16 b[NT];
        #pragma unroll
        for (int j = 0; j < NT; ++j) {
            const _Float16* bp = Bw + (size_t)(n0 + j * 16 + row) * K + k0 + hi * 8;
            b[j].h[0] = *(const v8h*)(bp);
            b[j].h[1] = *(const v8h*)(bp + 16);
        }

        #pragma unroll
        for (int j = 0; j < NT; ++j)
            acc[j] = __builtin_amdgcn_wmma_f32_16x16x32_f16(
                         /*neg_a=*/false, a.v, /*neg_b=*/false, b[j].v,
                         /*c_mod=*/(short)0, acc[j],
                         /*reuse_a=*/false, /*reuse_b=*/false);
    }

    if (mode == 1) {
        #pragma unroll
        for (int j = 0; j < NT; ++j)
            #pragma unroll
            for (int v = 0; v < 8; ++v)
                Cf32[(size_t)(m0 + hi * 8 + v) * ldc + (n0 + j * 16 + row)] = acc[j][v];
    } else {
        #pragma unroll
        for (int j = 0; j < NT; ++j)
            #pragma unroll
            for (int v = 0; v < 8; ++v)
                Cf16[(size_t)(m0 + hi * 8 + v) * ldc + (n0 + j * 16 + row)] =
                    (_Float16)acc[j][v];
    }
}

// ---------------------------------------------------------------------------
// Pack f32 weight [N,K] -> f16 [Npad,K] with zero padding rows.
// ---------------------------------------------------------------------------
__global__ __launch_bounds__(256)
void pack_w_f16(const float* __restrict__ src, _Float16* __restrict__ dst,
                int N, int Npad, int K)
{
    int i = blockIdx.x * blockDim.x + threadIdx.x;
    if (i >= Npad * K) return;
    int n = i / K, k = i - n * K;
    dst[i] = (n < N) ? (_Float16)src[n * K + k] : (_Float16)0.0f;
}

// ---------------------------------------------------------------------------
// Embedding gather: h[b,l,:] = embed[x[b,l],:]
// ---------------------------------------------------------------------------
__global__ __launch_bounds__(256)
void embed_kernel(const int* __restrict__ x, const float* __restrict__ embed,
                  float* __restrict__ h, int total)
{
    int i = blockIdx.x * blockDim.x + threadIdx.x;
    if (i >= total) return;
    int tok = x[i >> 6];
    h[i] = embed[tok * HID + (i & (HID - 1))];
}

// ---------------------------------------------------------------------------
// RMSNorm over HID=64, emit f16. One row per wave (2 floats / lane).
// ---------------------------------------------------------------------------
__global__ __launch_bounds__(256)
void rmsnorm_f16_kernel(const float* __restrict__ h, const float* __restrict__ w,
                        _Float16* __restrict__ out, int rows)
{
    const int lane = threadIdx.x & 31;
    const int wave = threadIdx.x >> 5;
    const int row  = blockIdx.x * 8 + wave;
    if (row >= rows) return;

    float2 v = ((const float2*)(h + (size_t)row * HID))[lane];
    float ss = v.x * v.x + v.y * v.y;
    #pragma unroll
    for (int m = 16; m >= 1; m >>= 1) ss += __shfl_xor(ss, m, 32);
    float r = rsqrtf(ss * (1.0f / HID) + 1e-5f);
    out[(size_t)row * HID + 2 * lane]     = (_Float16)(v.x * r * w[2 * lane]);
    out[(size_t)row * HID + 2 * lane + 1] = (_Float16)(v.y * r * w[2 * lane + 1]);
}

// ---------------------------------------------------------------------------
// Causal depthwise conv (K=4) + bias + silu on the first 128 cols of proj.
// ---------------------------------------------------------------------------
__global__ __launch_bounds__(256)
void conv_silu_kernel(const _Float16* __restrict__ proj,
                      const float* __restrict__ cw, const float* __restrict__ cb,
                      _Float16* __restrict__ hs, int total)
{
    int i = blockIdx.x * blockDim.x + threadIdx.x;
    if (i >= total) return;
    const int d = i & (DI - 1);
    const int m = i >> 7;
    const int l = m & (SEQ - 1);

    float acc = cb[d];
    #pragma unroll
    for (int j = 0; j < KC; ++j) {
        int ls = l - (KC - 1) + j;
        if (ls >= 0)
            acc += (float)proj[(size_t)(m - (KC - 1) + j) * (2 * DI) + d] * cw[d * KC + j];
    }
    float s = acc * (1.0f / (1.0f + __expf(-acc)));   // silu
    hs[i] = (_Float16)s;
}

// ---------------------------------------------------------------------------
// Fused selective-scan: dt-projection + softplus + 16-state recurrence +
// C-readout + D-skip + gate silu. One block per batch, one thread per DI
// channel. B/C rows staged through LDS in 16-step chunks.
// ssmp layout [M,48]: cols 0-3 dt_raw, 4-19 B, 20-35 C.
// ---------------------------------------------------------------------------
__global__ __launch_bounds__(128)
void scan_kernel(const _Float16* __restrict__ ssmp,
                 const _Float16* __restrict__ hs,
                 const _Float16* __restrict__ proj,      // gate at cols 128..255
                 const float* __restrict__ dtw, const float* __restrict__ dtb,
                 const float* __restrict__ A_log, const float* __restrict__ D_skip,
                 _Float16* __restrict__ y)
{
    const int b = blockIdx.x;
    const int d = threadIdx.x;

    float A[ST];
    #pragma unroll
    for (int n = 0; n < ST; ++n) A[n] = -__expf(A_log[d * ST + n]);
    const float w0 = dtw[d * RK + 0], w1 = dtw[d * RK + 1];
    const float w2 = dtw[d * RK + 2], w3 = dtw[d * RK + 3];
    const float bias = dtb[d];
    const float dsk  = D_skip[d];

    float st[ST];
    #pragma unroll
    for (int n = 0; n < ST; ++n) st[n] = 0.0f;

    __shared__ float sR[16][36];
    for (int l0 = 0; l0 < SEQ; l0 += 16) {
        __syncthreads();
        for (int i = threadIdx.x; i < 16 * 36; i += 128) {
            int ll = i / 36, c = i - ll * 36;
            sR[ll][c] = (float)ssmp[(size_t)(b * SEQ + l0 + ll) * 48 + c];
        }
        __syncthreads();
        for (int lc = 0; lc < 16; ++lc) {
            const size_t m = (size_t)b * SEQ + l0 + lc;
            float raw = sR[lc][0] * w0 + sR[lc][1] * w1 +
                        sR[lc][2] * w2 + sR[lc][3] * w3 + bias;
            float dt = (raw > 20.0f) ? raw : log1pf(__expf(raw));
            float hv = (float)hs[m * DI + d];
            float yv = 0.0f;
            #pragma unroll
            for (int n = 0; n < ST; ++n) {
                float dA = __expf(dt * A[n]);
                st[n] = dA * st[n] + (dt * sR[lc][RK + n]) * hv;
                yv += st[n] * sR[lc][RK + ST + n];
            }
            float g  = (float)proj[m * (2 * DI) + DI + d];
            float sg = g * (1.0f / (1.0f + __expf(-g)));
            yv = (yv + hv * dsk) * sg;
            y[m * DI + d] = (_Float16)yv;
        }
    }
}

// ---------------------------------------------------------------------------
// Final rmsnorm + max-pool over seq + 3-way classifier. One block per batch.
// ---------------------------------------------------------------------------
__global__ __launch_bounds__(64)
void final_kernel(const float* __restrict__ h, const float* __restrict__ nfw,
                  const float* __restrict__ bw, const float* __restrict__ bb,
                  float* __restrict__ out)
{
    const int b = blockIdx.x, t = threadIdx.x;
    const int lane = t & 31, wave = t >> 5;
    __shared__ float lds[2];
    __shared__ float pooled[HID];

    const float w = nfw[t];
    float mx = -3.4e38f;
    for (int l = 0; l < SEQ; ++l) {
        float v = h[((size_t)b * SEQ + l) * HID + t];
        float ss = v * v;
        #pragma unroll
        for (int m = 16; m >= 1; m >>= 1) ss += __shfl_xor(ss, m, 32);
        if (lane == 0) lds[wave] = ss;
        __syncthreads();
        float tot = lds[0] + lds[1];
        float hv = v * rsqrtf(tot * (1.0f / HID) + 1e-5f) * w;
        mx = fmaxf(mx, hv);
        __syncthreads();
    }
    pooled[t] = mx;
    __syncthreads();
    if (t < NCLS) {
        float acc = bb[t];
        for (int k = 0; k < HID; ++k) acc += pooled[k] * bw[t * HID + k];
        out[b * NCLS + t] = acc;
    }
}

// ---------------------------------------------------------------------------
// Host orchestration
// ---------------------------------------------------------------------------
extern "C" void kernel_launch(void* const* d_in, const int* in_sizes, int n_in,
                              void* d_out, int out_size, void* d_ws, size_t ws_size,
                              hipStream_t stream)
{
    const int*   x        = (const int*)  d_in[0];
    const float* embed    = (const float*)d_in[1];
    const float* in_w     = (const float*)d_in[2];   // [2,256,64]
    const float* conv_w   = (const float*)d_in[3];   // [2,128,4]
    const float* conv_b   = (const float*)d_in[4];   // [2,128]
    const float* x_w      = (const float*)d_in[5];   // [2,36,128]
    const float* dt_w     = (const float*)d_in[6];   // [2,128,4]
    const float* dt_b     = (const float*)d_in[7];   // [2,128]
    const float* A_log    = (const float*)d_in[8];   // [2,128,16]
    const float* D_skip   = (const float*)d_in[9];   // [2,128]
    const float* out_w    = (const float*)d_in[10];  // [2,64,128]
    const float* rms_w    = (const float*)d_in[11];  // [2,64]
    const float* norm_f_w = (const float*)d_in[12];  // [64]
    const float* bind_w   = (const float*)d_in[13];  // [3,64]
    const float* bind_b   = (const float*)d_in[14];  // [3]
    float* out = (float*)d_out;

    const int M = MROWS;
    char* ws = (char*)d_ws;
    size_t off = 0;
    auto alloc = [&](size_t bytes) -> void* {
        void* p = ws + off;
        off = (off + bytes + 255) & ~(size_t)255;
        return p;
    };

    float*     h    = (float*)    alloc((size_t)M * HID * 4);       // residual stream
    _Float16*  hn   = (_Float16*) alloc((size_t)M * HID * 2);       // rmsnorm f16
    _Float16*  proj = (_Float16*) alloc((size_t)M * 2 * DI * 2);    // in_proj out
    _Float16*  hs   = (_Float16*) alloc((size_t)M * DI * 2);        // conv+silu out
    _Float16*  ssmp = (_Float16*) alloc((size_t)M * 48 * 2);        // x_proj out (padded)
    _Float16*  yb   = (_Float16*) alloc((size_t)M * DI * 2);        // scan out
    _Float16*  win  = (_Float16*) alloc((size_t)NLAYERS * 256 * 64 * 2);
    _Float16*  wx   = (_Float16*) alloc((size_t)NLAYERS * 48 * 128 * 2);
    _Float16*  wo   = (_Float16*) alloc((size_t)NLAYERS * 64 * 128 * 2);

    // weight packing (f32 -> f16, pad x_proj N 36->48)
    for (int l = 0; l < NLAYERS; ++l) {
        pack_w_f16<<<(256 * 64 + 255) / 256, 256, 0, stream>>>(
            in_w + (size_t)l * 256 * 64, win + (size_t)l * 256 * 64, 256, 256, 64);
        pack_w_f16<<<(48 * 128 + 255) / 256, 256, 0, stream>>>(
            x_w + (size_t)l * 36 * 128, wx + (size_t)l * 48 * 128, 36, 48, 128);
        pack_w_f16<<<(64 * 128 + 255) / 256, 256, 0, stream>>>(
            out_w + (size_t)l * 64 * 128, wo + (size_t)l * 64 * 128, 64, 64, 128);
    }

    // embedding
    embed_kernel<<<(M * HID + 255) / 256, 256, 0, stream>>>(x, embed, h, M * HID);

    const int mTiles = M / 16;                                  // 4096
    for (int l = 0; l < NLAYERS; ++l) {
        // rmsnorm -> f16
        rmsnorm_f16_kernel<<<M / 8, 256, 0, stream>>>(h, rms_w + l * HID, hn, M);

        // in_proj: [M,64] x [256,64]^T -> f16 [M,256]; 4 n-groups of 4 tiles
        wmma_gemm_f16<4><<<mTiles * 4 / 8, 256, 0, stream>>>(
            hn, win + (size_t)l * 256 * 64, proj, nullptr, M, 64, 256, 4, 0);

        // depthwise causal conv + silu
        conv_silu_kernel<<<(M * DI + 255) / 256, 256, 0, stream>>>(
            proj, conv_w + l * DI * KC, conv_b + l * DI, hs, M * DI);

        // x_proj: [M,128] x [48,128]^T -> f16 [M,48]; one 3-tile strip
        wmma_gemm_f16<3><<<mTiles / 8, 256, 0, stream>>>(
            hs, wx + (size_t)l * 48 * 128, ssmp, nullptr, M, 128, 48, 1, 0);

        // fused dt + selective scan + gate
        scan_kernel<<<BATCH, DI, 0, stream>>>(
            ssmp, hs, proj, dt_w + l * DI * RK, dt_b + l * DI,
            A_log + l * DI * ST, D_skip + l * DI, yb);

        // out_proj with fused residual: h += [M,128] x [64,128]^T; one 4-tile strip
        wmma_gemm_f16<4><<<mTiles / 8, 256, 0, stream>>>(
            yb, wo + (size_t)l * 64 * 128, nullptr, h, M, 128, 64, 1, 1);
    }

    // final norm + maxpool + classifier
    final_kernel<<<BATCH, HID, 0, stream>>>(h, norm_f_w, bind_w, bind_b, out);
}
